// DynamicAttention1_40827959116533
// MI455X (gfx1250) — matrix operand order repaired and checked by reading, not compiled, not run on hardware
//
#include <hip/hip_runtime.h>

// ---------------------------------------------------------------------------
// CDNA5 (gfx1250) WMMA implementation of DynamicAttention1.
// Key algebraic fact: score(s,t) = (qs[s]-qt[t])/32 is separable, so the joint
// (S*T) softmax factorizes into two independent 1-D softmaxes (marginals):
//   ws = softmax_s(qs/32),  wt = softmax_t(-qt/32).
// Everything else is GEMMs, done in bf16 WMMA (f32 accumulate).
// ---------------------------------------------------------------------------

typedef __attribute__((ext_vector_type(16))) __bf16        v16bf;
typedef __attribute__((ext_vector_type(8)))  float         v8f;
typedef __attribute__((ext_vector_type(4)))  unsigned int  v4u;

using u16 = unsigned short;

// Problem constants (fixed by the reference).
constexpr int BB = 8;      // batch
constexpr int LL = 128;    // query length
constexpr int SS = 100;    // src length (valid)
constexpr int SP = 128;    // src/trg length padded to multiple of 32
constexpr int QD = 512;    // query dim
constexpr int FD = 2048;   // feat dim
constexpr int HD = 512;    // hidden dim
constexpr int OD = 512;    // out dim
constexpr int XK = 1024;   // concat dim (QD + HD)

__device__ __forceinline__ u16 f2bf(float f) {
    unsigned int u = __float_as_uint(f);
    u += 0x7fffu + ((u >> 16) & 1u);   // round-to-nearest-even truncation
    return (u16)(u >> 16);
}

// A-operand fragment (16x32 bf16, M x K): lane L<16 holds row M=L with
// K {k0..k0+7, k0+16..k0+23}; lane L>=16 holds row M=L-16 with
// K {k0+8..k0+15, k0+24..k0+31}.  (ISA 7.12.2, 16-bit A matrix)
__device__ __forceinline__ v16bf load_frag_a(const u16* __restrict__ base, int ld,
                                             int row0, int k0, int lane) {
    const int half = lane >> 4;
    const int r    = lane & 15;
    const u16* p = base + (long)(row0 + r) * ld + k0 + half * 8;
    union { v4u u[2]; v16bf v; } f;
    f.u[0] = *(const v4u*)(p);        // K +0..7   (relative to half)
    f.u[1] = *(const v4u*)(p + 16);   // K +16..23
    return f.v;
}

// B-operand fragment (32x16 bf16, K x N) supplied as rows of B^T (N x K,
// K-contiguous): lane L<16 holds column N=L with K {k0..k0+7, k0+16..k0+23};
// lane L>=16 holds column N=L-16 with K {k0+8..k0+15, k0+24..k0+31}.
__device__ __forceinline__ v16bf load_frag_b(const u16* __restrict__ baseT, int ld,
                                             int col0, int k0, int lane) {
    const int half = lane >> 4;
    const int n    = lane & 15;
    const u16* p = baseT + (long)(col0 + n) * ld + k0 + half * 8;
    union { v4u u[2]; v16bf v; } f;
    f.u[0] = *(const v4u*)(p);        // K +0..7   (relative to half)
    f.u[1] = *(const v4u*)(p + 16);   // K +16..23
    return f.v;
}

__device__ __forceinline__ v8f wmma_bf16(v16bf a, v16bf b, v8f c) {
    // v_wmma_f32_16x16x32_bf16 : D = A(16x32) * B(32x16) + C(16x16 f32)
    return __builtin_amdgcn_wmma_f32_16x16x32_bf16(
        /*neg_a=*/false, a, /*neg_b=*/false, b,
        /*c_mod=*/(short)0, c, /*reuse_a=*/false, /*reuse_b=*/false);
}

// ---------------------------------------------------------------------------
// Generic TN GEMM:  C[b] = alpha * (A[b] @ BT[b]^T  -  (DUAL? A2[b] @ B2T[b]^T : 0)) + bias
// A   : (M,K) bf16 row-major, leading dim lda, batch stride sA
// BT  : (N,K) bf16 row-major (i.e. B transposed), ldb, sB
// C   : f32 (Cf) or bf16 (Cb), leading dim ldc, batch stride sC
// CT  : optional bf16 transposed copy (N,M), ldct, sCT
// Block: 128 threads = 4 waves, each wave does a 32x32 tile (2x2 WMMA tiles);
// workgroup tile 64x64. All dims are exact multiples -> no bounds checks.
// ---------------------------------------------------------------------------
template<bool DUAL, bool HAS_BIAS, bool OUT_BF16, bool STORE_T>
__global__ __launch_bounds__(128) void da1_gemm_tn(
    const u16* __restrict__ A,   long sA,  int lda,
    const u16* __restrict__ BT,  long sB,  int ldb,
    const u16* __restrict__ A2,  long sA2,
    const u16* __restrict__ B2T, long sB2,
    float* __restrict__ Cf, u16* __restrict__ Cb, long sC, int ldc,
    u16* __restrict__ CT, long sCT, int ldct,
    const float* __restrict__ bias, float alpha, int K)
{
    const int lane = threadIdx.x & 31;
    const int wave = threadIdx.x >> 5;
    const int bz   = blockIdx.z;
    const int m0   = blockIdx.y * 64 + (wave >> 1) * 32;
    const int n0   = blockIdx.x * 64 + (wave & 1) * 32;

    const u16* Ab  = A  + (long)bz * sA;
    const u16* Bb  = BT + (long)bz * sB;
    const u16* A2b = DUAL ? (A2  + (long)bz * sA2) : nullptr;
    const u16* B2b = DUAL ? (B2T + (long)bz * sB2) : nullptr;

    v8f acc[2][2] = {};
    v8f dcc[2][2] = {};

    for (int k = 0; k < K; k += 32) {
        if (k + 32 < K) {  // prefetch next K tile (global_prefetch_b8)
            __builtin_prefetch(Ab + (long)(m0 + (lane & 15)) * lda + (k + 32), 0, 1);
            __builtin_prefetch(Bb + (long)(n0 + (lane & 15)) * ldb + (k + 32), 0, 1);
        }
        v16bf a0 = load_frag_a(Ab, lda, m0,      k, lane);
        v16bf a1 = load_frag_a(Ab, lda, m0 + 16, k, lane);
        v16bf b0 = load_frag_b(Bb, ldb, n0,      k, lane);
        v16bf b1 = load_frag_b(Bb, ldb, n0 + 16, k, lane);
        acc[0][0] = wmma_bf16(a0, b0, acc[0][0]);
        acc[0][1] = wmma_bf16(a0, b1, acc[0][1]);
        acc[1][0] = wmma_bf16(a1, b0, acc[1][0]);
        acc[1][1] = wmma_bf16(a1, b1, acc[1][1]);
        if (DUAL) {
            v16bf e0 = load_frag_a(A2b, lda, m0,      k, lane);
            v16bf e1 = load_frag_a(A2b, lda, m0 + 16, k, lane);
            v16bf f0 = load_frag_b(B2b, ldb, n0,      k, lane);
            v16bf f1 = load_frag_b(B2b, ldb, n0 + 16, k, lane);
            dcc[0][0] = wmma_bf16(e0, f0, dcc[0][0]);
            dcc[0][1] = wmma_bf16(e0, f1, dcc[0][1]);
            dcc[1][0] = wmma_bf16(e1, f0, dcc[1][0]);
            dcc[1][1] = wmma_bf16(e1, f1, dcc[1][1]);
        }
    }

    // C/D layout (ISA 7.12.2): element i of v8f is row (i + 8*(lane>=16)),
    // column (lane & 15) of the 16x16 tile.
    const int half = lane >> 4;
    const int col  = lane & 15;
    for (int mt = 0; mt < 2; ++mt) {
        for (int nt = 0; nt < 2; ++nt) {
            v8f a = acc[mt][nt];
            if (DUAL) a = a - dcc[mt][nt];
            const int gn = n0 + nt * 16 + col;
            const float bv = HAS_BIAS ? bias[gn] : 0.0f;
            for (int i = 0; i < 8; ++i) {
                const int gm = m0 + mt * 16 + i + 8 * half;
                const float v = a[i] * alpha + bv;
                if (OUT_BF16) Cb[(long)bz * sC + (long)gm * ldc + gn] = f2bf(v);
                else          Cf[(long)bz * sC + (long)gm * ldc + gn] = v;
                if (STORE_T)  CT[(long)bz * sCT + (long)gn * ldct + gm] = f2bf(v);
            }
        }
    }
}

// f32 -> bf16 convert with zero row padding; optional output leading dim.
__global__ void da1_pad_convert(const float* __restrict__ in, u16* __restrict__ out,
                                int rows_in, int rows_out, int cols, int out_ld, int nb)
{
    long idx = (long)blockIdx.x * blockDim.x + threadIdx.x;
    long total = (long)nb * rows_out * cols;
    if (idx >= total) return;
    int c = (int)(idx % cols);
    long t = idx / cols;
    int r = (int)(t % rows_out);
    int b = (int)(t / rows_out);
    float v = (r < rows_in) ? in[((long)b * rows_in + r) * cols + c] : 0.0f;
    out[((long)b * rows_out + r) * out_ld + c] = f2bf(v);
}

// (R,C) f32 -> (C,R) bf16
__global__ void da1_transpose_convert(const float* __restrict__ in, u16* __restrict__ out,
                                      int R, int C)
{
    long idx = (long)blockIdx.x * blockDim.x + threadIdx.x;
    long total = (long)R * C;
    if (idx >= total) return;
    int c = (int)(idx % C);
    int r = (int)(idx / C);
    out[(long)c * R + r] = f2bf(in[(long)r * C + c]);
}

// Wave32 row softmax over `valid` entries; writes bf16, zero in padded tail.
// blockIdx.y == 0 : ws = softmax(qs), blockIdx.y == 1 : wt = softmax(qtm)
// (the -1/32 score sign is folded into qtm's GEMM alpha).
__global__ void da1_softmax_rows(const float* __restrict__ qs, const float* __restrict__ qtm,
                                 u16* __restrict__ ws, u16* __restrict__ wt,
                                 int valid, int padded)
{
    const int row = blockIdx.x;
    const float* p = (blockIdx.y ? qtm : qs) + (long)row * padded;
    u16* o = (blockIdx.y ? wt : ws) + (long)row * padded;
    const int lane = threadIdx.x;

    float v[4];
    float m = -3.0e38f;
    for (int i = 0; i < 4; ++i) {
        int idx = lane + 32 * i;
        v[i] = (idx < valid) ? p[idx] : -3.0e38f;
        m = fmaxf(m, v[i]);
    }
    for (int off = 16; off; off >>= 1) m = fmaxf(m, __shfl_xor(m, off, 32));

    float e[4];
    float s = 0.0f;
    for (int i = 0; i < 4; ++i) {
        int idx = lane + 32 * i;
        e[i] = (idx < valid) ? __expf(v[i] - m) : 0.0f;
        s += e[i];
    }
    for (int off = 16; off; off >>= 1) s += __shfl_xor(s, off, 32);
    const float inv = 1.0f / s;

    for (int i = 0; i < 4; ++i) {
        int idx = lane + 32 * i;
        o[idx] = f2bf((idx < valid) ? e[i] * inv : 0.0f);
    }
}

extern "C" void kernel_launch(void* const* d_in, const int* in_sizes, int n_in,
                              void* d_out, int out_size, void* d_ws, size_t ws_size,
                              hipStream_t stream) {
    (void)in_sizes; (void)n_in; (void)out_size; (void)ws_size;

    const float* query = (const float*)d_in[0];  // (8,128,512)
    const float* src   = (const float*)d_in[1];  // (8,100,2048)
    const float* trg   = (const float*)d_in[2];  // (8,100,2048)
    const float* Wq    = (const float*)d_in[3];  // (512,512)
    const float* bq    = (const float*)d_in[4];  // (512)
    const float* Ws    = (const float*)d_in[5];  // (2048,512)
    const float* bs    = (const float*)d_in[6];  // (512)
    const float* Wo    = (const float*)d_in[7];  // (1024,512)
    const float* bo    = (const float*)d_in[8];  // (512)
    float* out = (float*)d_out;                  // (8,128,512)

    // ---- workspace carve-out (bf16 unless noted) ----
    char* w = (char*)d_ws;
    size_t off = 0;
    auto carve = [&](size_t bytes) {
        void* p = w + off;
        off = (off + bytes + 255) & ~(size_t)255;
        return p;
    };
    u16*   srcp  = (u16*)carve((size_t)BB * SP * FD * 2);   // padded src
    u16*   trgp  = (u16*)carve((size_t)BB * SP * FD * 2);   // padded trg
    u16*   xcat  = (u16*)carve((size_t)BB * LL * XK * 2);   // [query | ctx]
    u16*   WqT   = (u16*)carve((size_t)HD * QD * 2);        // (HD,QD)
    u16*   WsT   = (u16*)carve((size_t)HD * FD * 2);        // (HD,FD)
    u16*   WoT   = (u16*)carve((size_t)OD * XK * 2);        // (OD,XK)
    u16*   qb    = (u16*)carve((size_t)BB * LL * HD * 2);   // q
    u16*   skey  = (u16*)carve((size_t)BB * SP * HD * 2);
    u16*   tkey  = (u16*)carve((size_t)BB * SP * HD * 2);
    u16*   skeyT = (u16*)carve((size_t)BB * HD * SP * 2);
    u16*   tkeyT = (u16*)carve((size_t)BB * HD * SP * 2);
    float* qsf   = (float*)carve((size_t)BB * LL * SP * 4); //  q.skey/32
    float* qtmf  = (float*)carve((size_t)BB * LL * SP * 4); // -q.tkey/32
    u16*   wsb   = (u16*)carve((size_t)BB * LL * SP * 2);
    u16*   wtb   = (u16*)carve((size_t)BB * LL * SP * 2);

    const dim3 blk(128);
    const float inv_c = 1.0f / 32.0f;          // 1/sqrt(2*HID)
    const float inv_r2 = 0.70710678118654752f; // 1/sqrt(2)

    // ---- prep: convert/pad activations, transpose weights (all -> bf16) ----
    {
        long tot = (long)BB * SP * FD;
        da1_pad_convert<<<dim3((unsigned)((tot + 255) / 256)), 256, 0, stream>>>(src, srcp, SS, SP, FD, FD, BB);
        da1_pad_convert<<<dim3((unsigned)((tot + 255) / 256)), 256, 0, stream>>>(trg, trgp, SS, SP, FD, FD, BB);
        long tq = (long)BB * LL * QD;
        da1_pad_convert<<<dim3((unsigned)((tq + 255) / 256)), 256, 0, stream>>>(query, xcat, LL, LL, QD, XK, BB);
        da1_transpose_convert<<<dim3((QD * HD + 255) / 256), 256, 0, stream>>>(Wq, WqT, QD, HD);
        da1_transpose_convert<<<dim3((FD * HD + 255) / 256), 256, 0, stream>>>(Ws, WsT, FD, HD);
        da1_transpose_convert<<<dim3((XK * OD + 255) / 256), 256, 0, stream>>>(Wo, WoT, XK, OD);
    }

    // ---- keys: skey/tkey = pad(src/trg) @ Ws + bs ; also emit transposed bf16 copy ----
    da1_gemm_tn<false, true, true, true><<<dim3(HD / 64, SP / 64, BB), blk, 0, stream>>>(
        srcp, (long)SP * FD, FD,  WsT, 0L, FD,
        nullptr, 0L, nullptr, 0L,
        nullptr, skey, (long)SP * HD, HD,
        skeyT, (long)HD * SP, SP,
        bs, 1.0f, FD);
    da1_gemm_tn<false, true, true, true><<<dim3(HD / 64, SP / 64, BB), blk, 0, stream>>>(
        trgp, (long)SP * FD, FD,  WsT, 0L, FD,
        nullptr, 0L, nullptr, 0L,
        nullptr, tkey, (long)SP * HD, HD,
        tkeyT, (long)HD * SP, SP,
        bs, 1.0f, FD);

    // ---- q = query @ Wq + bq (A read from xcat's first half, lda = XK) ----
    da1_gemm_tn<false, true, true, false><<<dim3(HD / 64, LL / 64, BB), blk, 0, stream>>>(
        xcat, (long)LL * XK, XK,  WqT, 0L, QD,
        nullptr, 0L, nullptr, 0L,
        nullptr, qb, (long)LL * HD, HD,
        nullptr, 0L, 0,
        bq, 1.0f, QD);

    // ---- scores: qs = (q.skey)/32 ; qtm = -(q.tkey)/32  (B operand = key rows, K-contig) ----
    da1_gemm_tn<false, false, false, false><<<dim3(SP / 64, LL / 64, BB), blk, 0, stream>>>(
        qb, (long)LL * HD, HD,  skey, (long)SP * HD, HD,
        nullptr, 0L, nullptr, 0L,
        qsf, nullptr, (long)LL * SP, SP,
        nullptr, 0L, 0,
        nullptr, inv_c, HD);
    da1_gemm_tn<false, false, false, false><<<dim3(SP / 64, LL / 64, BB), blk, 0, stream>>>(
        qb, (long)LL * HD, HD,  tkey, (long)SP * HD, HD,
        nullptr, 0L, nullptr, 0L,
        qtmf, nullptr, (long)LL * SP, SP,
        nullptr, 0L, 0,
        nullptr, -inv_c, HD);

    // ---- factored softmax marginals: ws = softmax(qs), wt = softmax(qtm) ----
    da1_softmax_rows<<<dim3(BB * LL, 2), 32, 0, stream>>>(qsf, qtmf, wsb, wtb, SS, SP);

    // ---- ctx = (ws @ skey - wt @ tkey)/sqrt(2), written into xcat[:, :, 512:1024] ----
    da1_gemm_tn<true, false, true, false><<<dim3(HD / 64, LL / 64, BB), blk, 0, stream>>>(
        wsb, (long)LL * SP, SP,  skeyT, (long)HD * SP, SP,
        wtb, (long)LL * SP,      tkeyT, (long)HD * SP,
        nullptr, xcat + QD, (long)LL * XK, XK,
        nullptr, 0L, 0,
        nullptr, inv_r2, SP);

    // ---- out = [query, ctx] @ Wo + bo  (f32) ----
    da1_gemm_tn<false, true, false, false><<<dim3(OD / 64, LL / 64, BB), blk, 0, stream>>>(
        xcat, (long)LL * XK, XK,  WoT, 0L, XK,
        nullptr, 0L, nullptr, 0L,
        out, nullptr, (long)LL * OD, OD,
        nullptr, 0L, 0,
        bo, 1.0f, XK);
}
